// LSTM_27779848471481
// MI455X (gfx1250) — matrix operand order repaired and checked by reading, not compiled, not run on hardware
//
#include <hip/hip_runtime.h>

// ---------------------------------------------------------------------------
// 2-layer LSTM + classifier for MI455X (gfx1250), bf16 WMMA with f32 accum.
// B=32, T=512, I=256, H=1024, O=256.
// Step kernel: one wave owns BOTH 16-row batch tiles and all 4 gates for a
// 16-wide hidden slice -> every B (weight) fragment feeds 2 WMMAs.
// ---------------------------------------------------------------------------

#define DEV __device__ __forceinline__

typedef __attribute__((ext_vector_type(16))) __bf16 v16bf;
typedef __attribute__((ext_vector_type(8)))  __bf16 v8bf;
typedef __attribute__((ext_vector_type(8)))  float  v8f;

union FragA { v16bf v; v8bf h[2]; };

constexpr int Bsz  = 32;
constexpr int Tlen = 512;
constexpr int Isz  = 256;
constexpr int Hsz  = 1024;
constexpr int Osz  = 256;

DEV unsigned short f32_to_bf16(float x) {
  union { float f; unsigned int u; } c; c.f = x;
  unsigned int u = c.u + 0x7FFFu + ((c.u >> 16) & 1u);   // round-to-nearest-even
  return (unsigned short)(u >> 16);
}

DEV float sigmoid_f(float x) { return 1.0f / (1.0f + __expf(-x)); }

#define WMMA_BF16(A, B, C) \
  __builtin_amdgcn_wmma_f32_16x16x32_bf16(false, (A), false, (B), (short)0, (C), false, false)

// ------------------------------ utility kernels ----------------------------

__global__ void cvt_f32_bf16(const float* __restrict__ s,
                             unsigned short* __restrict__ d, int n) {
  int i = blockIdx.x * blockDim.x + threadIdx.x;
  if (i < n) d[i] = f32_to_bf16(s[i]);
}

__global__ void add_bias_f32(const float* __restrict__ a,
                             const float* __restrict__ b,
                             float* __restrict__ o, int n) {
  int i = blockIdx.x * blockDim.x + threadIdx.x;
  if (i < n) o[i] = a[i] + b[i];
}

__global__ void zero_u32(unsigned int* __restrict__ p, int n) {
  int i = blockIdx.x * blockDim.x + threadIdx.x;
  if (i < n) p[i] = 0u;
}

// ------------------------------ LSTM step kernel ---------------------------
// gates[32,4H] = [A0 | h_prev] @ [Wih ; Whh]^T + bias, then pointwise LSTM.
// Grid: 64 waves (H/16 hidden tiles); each wave: 2 batch tiles x 4 gates
// = 8 accumulators, B fragments reused across both batch tiles.
//
// WMMA operand layouts used here (wave32; A/B k order is this code's contiguous one):
//   A 16x32 bf16 : lane L -> row = L&15 ; khalf = L>>4 ;
//                  K = khalf*16+{0..7}  (v0..3)  and khalf*16+8+{0..7} (v4..7)
//   B 32x16 bf16 : lane L -> col = L&15 ; K = (L>>4)*16 + {0..15}  (one 32B load)
//   C/D f32 16x16: vgpr r, lane L -> row = r + 8*(L>>4), col = L&15
__global__ __launch_bounds__(128)
void lstm_step(const unsigned short* __restrict__ A0, int lda0, int K0,  // x-part
               const unsigned short* __restrict__ Aprev,                 // h_prev [32,H]
               const unsigned short* __restrict__ Wih,                   // [4H,K0] bf16
               const unsigned short* __restrict__ Whh,                   // [4H,H]  bf16
               const float* __restrict__ bias,                           // [4H] (b_ih+b_hh)
               float* __restrict__ c_state,                              // [32,H] f32
               unsigned short* __restrict__ h_out,                       // [32,H] bf16
               unsigned short* __restrict__ h_seq,                       // nullable
               long long seq_stride)
{
  const int lane = threadIdx.x & 31;
  const int j    = blockIdx.x * (blockDim.x >> 5) + (threadIdx.x >> 5);  // 0..63
  const int l16  = lane & 15;
  const int kh   = lane >> 4;
  const int hcol = j * 16 + l16;     // hidden unit / weight row this lane owns

  // acc[mt][gate]; mt=0 -> batch rows 0..15, mt=1 -> rows 16..31
  v8f acc00 = {}, acc01 = {}, acc02 = {}, acc03 = {};
  v8f acc10 = {}, acc11 = {}, acc12 = {}, acc13 = {};
  FragA a0, a1;

  // ---- x-part: K0 columns from A0 (row stride lda0), weights Wih ----
  {
    const __bf16* ap0 = (const __bf16*)A0 + (long long)l16 * lda0 + kh * 16;
    const __bf16* ap1 = ap0 + (long long)16 * lda0;
    const __bf16* bp  = (const __bf16*)Wih + (long long)hcol * K0 + kh * 16;
    const long long gstep = (long long)Hsz * K0;
    const int KB = K0 >> 5;
    for (int kb = 0; kb < KB; ++kb) {
      a0.h[0] = *(const v8bf*)(ap0 + kb * 32);
      a0.h[1] = *(const v8bf*)(ap0 + kb * 32 + 8);
      a1.h[0] = *(const v8bf*)(ap1 + kb * 32);
      a1.h[1] = *(const v8bf*)(ap1 + kb * 32 + 8);
      v16bf w0 = *(const v16bf*)(bp + kb * 32);
      v16bf w1 = *(const v16bf*)(bp + gstep     + kb * 32);
      v16bf w2 = *(const v16bf*)(bp + gstep * 2 + kb * 32);
      v16bf w3 = *(const v16bf*)(bp + gstep * 3 + kb * 32);
      acc00 = WMMA_BF16(a0.v, w0, acc00);  acc10 = WMMA_BF16(a1.v, w0, acc10);
      acc01 = WMMA_BF16(a0.v, w1, acc01);  acc11 = WMMA_BF16(a1.v, w1, acc11);
      acc02 = WMMA_BF16(a0.v, w2, acc02);  acc12 = WMMA_BF16(a1.v, w2, acc12);
      acc03 = WMMA_BF16(a0.v, w3, acc03);  acc13 = WMMA_BF16(a1.v, w3, acc13);
    }
  }

  // ---- recurrent part: H columns from h_prev, weights Whh ----
  {
    const __bf16* ap0 = (const __bf16*)Aprev + (long long)l16 * Hsz + kh * 16;
    const __bf16* ap1 = ap0 + (long long)16 * Hsz;
    const __bf16* bp  = (const __bf16*)Whh + (long long)hcol * Hsz + kh * 16;
    const long long gstep = (long long)Hsz * Hsz;
    const int KB = Hsz >> 5;
    for (int kb = 0; kb < KB; ++kb) {
      a0.h[0] = *(const v8bf*)(ap0 + kb * 32);
      a0.h[1] = *(const v8bf*)(ap0 + kb * 32 + 8);
      a1.h[0] = *(const v8bf*)(ap1 + kb * 32);
      a1.h[1] = *(const v8bf*)(ap1 + kb * 32 + 8);
      v16bf w0 = *(const v16bf*)(bp + kb * 32);
      v16bf w1 = *(const v16bf*)(bp + gstep     + kb * 32);
      v16bf w2 = *(const v16bf*)(bp + gstep * 2 + kb * 32);
      v16bf w3 = *(const v16bf*)(bp + gstep * 3 + kb * 32);
      acc00 = WMMA_BF16(a0.v, w0, acc00);  acc10 = WMMA_BF16(a1.v, w0, acc10);
      acc01 = WMMA_BF16(a0.v, w1, acc01);  acc11 = WMMA_BF16(a1.v, w1, acc11);
      acc02 = WMMA_BF16(a0.v, w2, acc02);  acc12 = WMMA_BF16(a1.v, w2, acc12);
      acc03 = WMMA_BF16(a0.v, w3, acc03);  acc13 = WMMA_BF16(a1.v, w3, acc13);
    }
  }

  // ---- fused pointwise LSTM: i,f,g,o live in the same lane/vgpr slot ----
  const float bi = bias[hcol];
  const float bf = bias[Hsz + hcol];
  const float bg = bias[2 * Hsz + hcol];
  const float bo = bias[3 * Hsz + hcol];

#pragma unroll
  for (int mt = 0; mt < 2; ++mt) {
    const v8f& gi = mt ? acc10 : acc00;
    const v8f& gf = mt ? acc11 : acc01;
    const v8f& gg = mt ? acc12 : acc02;
    const v8f& go = mt ? acc13 : acc03;
#pragma unroll
    for (int r = 0; r < 8; ++r) {
      const int brow = mt * 16 + r + 8 * kh;          // C/D layout row (batch)
      const long long idx = (long long)brow * Hsz + hcol;
      float iv = sigmoid_f(gi[r] + bi);
      float fv = sigmoid_f(gf[r] + bf);
      float gv = tanhf(gg[r] + bg);
      float ov = sigmoid_f(go[r] + bo);
      float cn = fv * c_state[idx] + iv * gv;
      c_state[idx] = cn;
      float hv = ov * tanhf(cn);
      unsigned short hb = f32_to_bf16(hv);
      h_out[idx] = hb;
      if (h_seq) h_seq[(long long)brow * seq_stride + hcol] = hb;
    }
  }
}

// ------------------------------ classifier GEMM ----------------------------
// C[M,N] = A[M,K](bf16,row-major) * Bw[N,K](bf16,row-major)^T + bias[N], f32 out.
// Each wave: 64 rows (4 M-tiles) x 16 cols; B fragment reused across 4 WMMAs.
__global__ __launch_bounds__(256)
void gemm_bf16_f32(const unsigned short* __restrict__ A,
                   const unsigned short* __restrict__ Bw,
                   const float* __restrict__ bias,
                   float* __restrict__ C, int M, int N, int K)
{
  const int lane = threadIdx.x & 31;
  const int tile = blockIdx.x * (blockDim.x >> 5) + (threadIdx.x >> 5);
  const int ntiles = N >> 4;
  const int mg  = tile / ntiles;            // 64-row group
  const int ntv = tile - mg * ntiles;
  if (mg * 64 >= M) return;                 // wave-uniform guard (EXEC stays full)

  const int l16 = lane & 15;
  const int kh  = lane >> 4;

  const __bf16* ap = (const __bf16*)A  + (long long)(mg * 64 + l16) * K + kh * 16;
  const __bf16* bp = (const __bf16*)Bw + (long long)(ntv * 16 + l16) * K + kh * 16;
  const long long rstep = (long long)16 * K;

  v8f c0 = {}, c1 = {}, c2 = {}, c3 = {};
  FragA a0, a1, a2, a3;
  const int KB = K >> 5;
  for (int kb = 0; kb < KB; ++kb) {
    a0.h[0] = *(const v8bf*)(ap + kb * 32);
    a0.h[1] = *(const v8bf*)(ap + kb * 32 + 8);
    a1.h[0] = *(const v8bf*)(ap + rstep + kb * 32);
    a1.h[1] = *(const v8bf*)(ap + rstep + kb * 32 + 8);
    a2.h[0] = *(const v8bf*)(ap + rstep * 2 + kb * 32);
    a2.h[1] = *(const v8bf*)(ap + rstep * 2 + kb * 32 + 8);
    a3.h[0] = *(const v8bf*)(ap + rstep * 3 + kb * 32);
    a3.h[1] = *(const v8bf*)(ap + rstep * 3 + kb * 32 + 8);
    v16bf w = *(const v16bf*)(bp + kb * 32);
    c0 = WMMA_BF16(a0.v, w, c0);
    c1 = WMMA_BF16(a1.v, w, c1);
    c2 = WMMA_BF16(a2.v, w, c2);
    c3 = WMMA_BF16(a3.v, w, c3);
  }

  const float bn = bias[ntv * 16 + l16];
#pragma unroll
  for (int mi = 0; mi < 4; ++mi) {
    const v8f& acc = mi == 0 ? c0 : mi == 1 ? c1 : mi == 2 ? c2 : c3;
#pragma unroll
    for (int r = 0; r < 8; ++r) {
      const int row = mg * 64 + mi * 16 + r + 8 * kh;
      C[(long long)row * N + ntv * 16 + l16] = acc[r] + bn;
    }
  }
}

// ------------------------------ launch -------------------------------------

extern "C" void kernel_launch(void* const* d_in, const int* in_sizes, int n_in,
                              void* d_out, int out_size, void* d_ws, size_t ws_size,
                              hipStream_t stream) {
  (void)in_sizes; (void)n_in; (void)out_size; (void)ws_size;

  const float* x    = (const float*)d_in[0];
  const float* Wih0 = (const float*)d_in[1];
  const float* Whh0 = (const float*)d_in[2];
  const float* bih0 = (const float*)d_in[3];
  const float* bhh0 = (const float*)d_in[4];
  const float* Wih1 = (const float*)d_in[5];
  const float* Whh1 = (const float*)d_in[6];
  const float* bih1 = (const float*)d_in[7];
  const float* bhh1 = (const float*)d_in[8];
  const float* Wc   = (const float*)d_in[9];
  const float* bc   = (const float*)d_in[10];

  // ---- carve workspace (all chunks are multiples of 256 bytes) ----
  char* ws = (char*)d_ws;
  auto carve = [&](size_t bytes) -> char* {
    char* p = ws; ws += (bytes + 255) & ~(size_t)255; return p;
  };
  unsigned short* xbf   = (unsigned short*)carve((size_t)Bsz * Tlen * Isz * 2);
  unsigned short* wih0b = (unsigned short*)carve((size_t)4 * Hsz * Isz * 2);
  unsigned short* whh0b = (unsigned short*)carve((size_t)4 * Hsz * Hsz * 2);
  unsigned short* wih1b = (unsigned short*)carve((size_t)4 * Hsz * Hsz * 2);
  unsigned short* whh1b = (unsigned short*)carve((size_t)4 * Hsz * Hsz * 2);
  unsigned short* wcb   = (unsigned short*)carve((size_t)Osz * Hsz * 2);
  float*          bias0 = (float*)carve((size_t)4 * Hsz * 4);
  float*          bias1 = (float*)carve((size_t)4 * Hsz * 4);
  // state block (zeroed each call): c0,c1 f32 + 4 h ping-pong buffers bf16
  char* stateBase = ws;
  float*          c0  = (float*)carve((size_t)Bsz * Hsz * 4);
  float*          c1  = (float*)carve((size_t)Bsz * Hsz * 4);
  unsigned short* h0a = (unsigned short*)carve((size_t)Bsz * Hsz * 2);
  unsigned short* h0b = (unsigned short*)carve((size_t)Bsz * Hsz * 2);
  unsigned short* h1a = (unsigned short*)carve((size_t)Bsz * Hsz * 2);
  unsigned short* h1b = (unsigned short*)carve((size_t)Bsz * Hsz * 2);
  size_t stateBytes = (size_t)(ws - stateBase);
  unsigned short* h2seq = (unsigned short*)carve((size_t)Bsz * Tlen * Hsz * 2);

  unsigned short* h0buf[2] = { h0a, h0b };
  unsigned short* h1buf[2] = { h1a, h1b };

  const int TPB = 256;
  auto nb = [&](long long n) { return (unsigned)((n + TPB - 1) / TPB); };

  // ---- precision conversion (fp32 -> bf16 bits) ----
  cvt_f32_bf16<<<nb((long long)Bsz * Tlen * Isz), TPB, 0, stream>>>(x, xbf, Bsz * Tlen * Isz);
  cvt_f32_bf16<<<nb((long long)4 * Hsz * Isz), TPB, 0, stream>>>(Wih0, wih0b, 4 * Hsz * Isz);
  cvt_f32_bf16<<<nb((long long)4 * Hsz * Hsz), TPB, 0, stream>>>(Whh0, whh0b, 4 * Hsz * Hsz);
  cvt_f32_bf16<<<nb((long long)4 * Hsz * Hsz), TPB, 0, stream>>>(Wih1, wih1b, 4 * Hsz * Hsz);
  cvt_f32_bf16<<<nb((long long)4 * Hsz * Hsz), TPB, 0, stream>>>(Whh1, whh1b, 4 * Hsz * Hsz);
  cvt_f32_bf16<<<nb((long long)Osz * Hsz), TPB, 0, stream>>>(Wc, wcb, Osz * Hsz);

  // combined gate biases (b_ih + b_hh) in f32
  add_bias_f32<<<nb(4 * Hsz), TPB, 0, stream>>>(bih0, bhh0, bias0, 4 * Hsz);
  add_bias_f32<<<nb(4 * Hsz), TPB, 0, stream>>>(bih1, bhh1, bias1, 4 * Hsz);

  // zero-init h/c state (harness poisons workspace)
  zero_u32<<<nb((long long)(stateBytes / 4)), TPB, 0, stream>>>((unsigned int*)stateBase,
                                                                (int)(stateBytes / 4));

  // ---- sequential recurrence: 512 steps x 2 layers ----
  for (int t = 0; t < Tlen; ++t) {
    const int pi = t & 1, po = (t + 1) & 1;
    // layer 0: x-part comes from xbf[b, t, :] (row stride T*I)
    lstm_step<<<16, 128, 0, stream>>>(
        xbf + (size_t)t * Isz, Tlen * Isz, Isz,
        h0buf[pi], wih0b, whh0b, bias0, c0, h0buf[po],
        (unsigned short*)nullptr, 0LL);
    // layer 1: x-part is layer-0's freshly written h (contiguous [32,H]);
    // also streams h2_t into the h2 sequence for the classifier.
    lstm_step<<<16, 128, 0, stream>>>(
        h0buf[po], Hsz, Hsz,
        h1buf[pi], wih1b, whh1b, bias1, c1, h1buf[po],
        h2seq + (size_t)t * Hsz, (long long)Tlen * Hsz);
  }

  // ---- classifier: [B*T, H] x [O, H]^T + b_c -> f32 d_out ----
  const int M = Bsz * Tlen;                       // 16384
  const int tiles = (M / 64) * (Osz >> 4);        // 4096 wave-tiles (64x16 each)
  gemm_bf16_f32<<<tiles / 8, 256, 0, stream>>>(h2seq, wcb, bc, (float*)d_out,
                                               M, Osz, Hsz);
}
